// SingleGroupProcessor_47132971107204
// MI455X (gfx1250) — compile-verified, matrix operand order checked by reading, not run on hardware
//
#include <hip/hip_runtime.h>
#include <math.h>

typedef __bf16 bf16;
typedef __attribute__((ext_vector_type(16))) __bf16 v16bf;
typedef __attribute__((ext_vector_type(8)))  float  v8f;
typedef __attribute__((ext_vector_type(4)))  int    ai4;

#define ADA_STRIDE 6144

#if __has_builtin(__builtin_amdgcn_global_load_async_to_lds_b128) && \
    __has_builtin(__builtin_amdgcn_s_wait_asynccnt)
#define USE_ASYNC_LDS 1
#endif

union FragU { uint4 u[2]; v16bf v; };

// CDNA5 16-bit A/B fragment: lanes 0-15 hold k = kb..kb+7 (elems 0-7) and
// kb+16..kb+23 (elems 8-15) where kb = (lane>>4)*8. Two 16B LDS loads.
__device__ __forceinline__ v16bf ld_frag(const bf16* p) {
  FragU f;
  f.u[0] = *reinterpret_cast<const uint4*>(p);
  f.u[1] = *reinterpret_cast<const uint4*>(p + 16);
  return f.v;
}

__device__ __forceinline__ v8f wmma_bf16(v16bf a, v16bf b, v8f c) {
  return __builtin_amdgcn_wmma_f32_16x16x32_bf16(false, a, false, b, (short)0, c, false, false);
}

// ---------------------------------------------------------------------------
// Weight convert + transpose: W[k][n] fp32 -> Wt[n][k] bf16
// ---------------------------------------------------------------------------
__global__ __launch_bounds__(256) void wconv_kernel(const float* __restrict__ W,
                                                    bf16* __restrict__ Wt,
                                                    int K, int N) {
  const int n = blockIdx.x * 256 + threadIdx.x;
  const int k = blockIdx.y;
  if (n < N) Wt[(size_t)n * K + k] = (bf16)W[(size_t)k * N + n];
}

// ---------------------------------------------------------------------------
// ada = silu(context) @ W_ada + b_ada   (8 x 6144, tiny -> VALU)
// ---------------------------------------------------------------------------
__global__ __launch_bounds__(256) void ada_kernel(const float* __restrict__ ctx,
                                                  const float* __restrict__ W,
                                                  const float* __restrict__ bias,
                                                  float* __restrict__ ada) {
  const int n = blockIdx.x * 256 + threadIdx.x;
  const int b = blockIdx.y;
  float s = bias[n];
  const float* c = ctx + (size_t)b * 1024;
  for (int k = 0; k < 1024; ++k) {
    float x = c[k];
    float sl = x / (1.0f + __expf(-x));
    s += sl * W[(size_t)k * ADA_STRIDE + n];
  }
  ada[(size_t)b * ADA_STRIDE + n] = s;
}

// ---------------------------------------------------------------------------
// RMSNorm + adaLN modulation -> bf16 (one block per token)
// ---------------------------------------------------------------------------
__global__ __launch_bounds__(256) void rmsnorm_mod_kernel(
    const float* __restrict__ x, const float* __restrict__ scale,
    const float* __restrict__ ada, int off_sh, int off_sc,
    bf16* __restrict__ out) {
  const int t = blockIdx.x;
  const int b = t >> 12;                 // 4096 tokens per batch
  const int tid = threadIdx.x;
  const float* row = x + (size_t)t * 1024;
  float v[4];
  float s = 0.f;
#pragma unroll
  for (int i = 0; i < 4; ++i) { v[i] = row[tid + i * 256]; s += v[i] * v[i]; }
#pragma unroll
  for (int m = 16; m >= 1; m >>= 1) s += __shfl_xor(s, m, 32);
  __shared__ float red[8];
  if ((tid & 31) == 0) red[tid >> 5] = s;
  __syncthreads();
  if (tid == 0) { float tt = 0.f; for (int k = 0; k < 8; ++k) tt += red[k]; red[0] = tt; }
  __syncthreads();
  const float inv = rsqrtf(red[0] * (1.0f / 1024.0f) + 1e-5f);
  const float* ab = ada + (size_t)b * ADA_STRIDE;
#pragma unroll
  for (int i = 0; i < 4; ++i) {
    int c = tid + i * 256;
    float y = v[i] * inv * scale[c];
    y = y * (1.0f + ab[off_sc + c]) + ab[off_sh + c];
    out[(size_t)t * 1024 + c] = (bf16)y;
  }
}

// ---------------------------------------------------------------------------
// RoPE on q and k in-place in the bf16 qkv buffer (position = token % 128)
// ---------------------------------------------------------------------------
__global__ __launch_bounds__(256) void rope_kernel(bf16* __restrict__ qkv) {
  const size_t idx = (size_t)blockIdx.x * 256 + threadIdx.x;   // 32768*8*64
  const int j = (int)(idx & 63);
  const int hh = (int)((idx >> 6) & 7);
  const size_t t = idx >> 9;
  const float p = (float)(t & 127);
  const float ln10k = 9.210340371976184f;
  const float f1 = __expf(-((float)(2 * (j >> 1)) * (1.0f / 128.0f)) * ln10k);
  const float f2 = __expf(-((float)(2 * (32 + (j >> 1))) * (1.0f / 128.0f)) * ln10k);
  const float a1 = p * f1, a2 = p * f2;
  const float c1 = __cosf(a1), s1 = __sinf(a1);
  const float c2 = __cosf(a2), s2 = __sinf(a2);
  bf16* q = qkv + t * 3072 + hh * 128;
  bf16* k = q + 1024;
  float q0 = (float)q[j], q1 = (float)q[j + 64];
  q[j]      = (bf16)(q0 * c1 - q1 * s1);
  q[j + 64] = (bf16)(q1 * c2 + q0 * s2);
  float k0 = (float)k[j], k1 = (float)k[j + 64];
  k[j]      = (bf16)(k0 * c1 - k1 * s1);
  k[j + 64] = (bf16)(k1 * c2 + k0 * s2);
}

// ---------------------------------------------------------------------------
// Generic 128x128-tile bf16 WMMA GEMM: C = A[MxK] @ Bt[NxK]^T + bias
// mode 0: out_bf = acc+bias
// mode 1: out_f  = resid + gate[batch][col]*(acc+bias)
// mode 2: out_bf = gelu_tanh(acc+bias)
// Async double-buffered global->LDS pipeline when the gfx1250 async builtins
// are available; register-staged pipeline otherwise.
// ---------------------------------------------------------------------------
#define LDT 40
#define TSZ (128 * LDT)

__global__ __launch_bounds__(256) void gemm_bf16_kernel(
    const bf16* __restrict__ A, const bf16* __restrict__ Bt,
    const float* __restrict__ bias,
    const float* __restrict__ resid, const float* __restrict__ gate,
    bf16* __restrict__ out_bf, float* __restrict__ out_f,
    int N, int K, int mode) {
  const int tid = threadIdx.x;
  const int lane = tid & 31;
  const int wave = tid >> 5;
  const int wm = wave >> 1;          // 0..3  (32-row strip)
  const int wn = wave & 1;           // 0..1  (64-col strip)
  const int hf = lane >> 4;
  const int rr = lane & 15;
  const int kb = hf * 8;
  const int mblk = blockIdx.y * 128;
  const int nblk = blockIdx.x * 128;
  const int lr = tid >> 1;
  const int lc = (tid & 1) * 16;

  const v8f vzero = {0.f, 0.f, 0.f, 0.f, 0.f, 0.f, 0.f, 0.f};
  v8f acc[2][4];
#pragma unroll
  for (int i = 0; i < 2; ++i)
#pragma unroll
    for (int j = 0; j < 4; ++j) acc[i][j] = vzero;

  const bf16* gA = A + (size_t)(mblk + lr) * K + lc;
  const bf16* gB = Bt + (size_t)(nblk + lr) * K + lc;
  const int nt = K >> 5;

#ifdef USE_ASYNC_LDS
  __shared__ bf16 As[2 * TSZ];
  __shared__ bf16 Bs[2 * TSZ];
  bf16* lA = As + lr * LDT + lc;
  bf16* lB = Bs + lr * LDT + lc;
  // each thread moves a 16-element (32B) slice of the A and B tiles as two
  // 16B async global->LDS transfers (builtin takes int4 as1/as3 pointers)
#define ASY(g, l) do {                                                        \
    __builtin_amdgcn_global_load_async_to_lds_b128(                            \
        (__attribute__((address_space(1))) ai4*)(g),                           \
        (__attribute__((address_space(3))) ai4*)(l), 0, 0);                    \
    __builtin_amdgcn_global_load_async_to_lds_b128(                            \
        (__attribute__((address_space(1))) ai4*)((g) + 8),                     \
        (__attribute__((address_space(3))) ai4*)((l) + 8), 0, 0);              \
  } while (0)

  ASY(gA, lA);
  ASY(gB, lB);
  for (int i = 0; i < nt; ++i) {
    const int cur = i & 1;
    __builtin_amdgcn_s_wait_asynccnt(0);   // my slices of tile i landed
    __syncthreads();                       // everyone's slices landed; buf cur^1 free
    if (i + 1 < nt) {
      const int kt = (i + 1) << 5;
      ASY(gA + kt, lA + (cur ^ 1) * TSZ);
      ASY(gB + kt, lB + (cur ^ 1) * TSZ);
    }
    const bf16* Ab = As + cur * TSZ;
    const bf16* Bb = Bs + cur * TSZ;
    v16bf af[2], bfg[4];
#pragma unroll
    for (int ii = 0; ii < 2; ++ii)
      af[ii] = ld_frag(Ab + (wm * 32 + ii * 16 + rr) * LDT + kb);
#pragma unroll
    for (int j = 0; j < 4; ++j)
      bfg[j] = ld_frag(Bb + (wn * 64 + j * 16 + rr) * LDT + kb);
#pragma unroll
    for (int ii = 0; ii < 2; ++ii)
#pragma unroll
      for (int j = 0; j < 4; ++j)
        acc[ii][j] = wmma_bf16(af[ii], bfg[j], acc[ii][j]);
  }
#undef ASY
#else
  __shared__ bf16 As[TSZ];
  __shared__ bf16 Bs[TSZ];
  uint4 a0 = *(const uint4*)(gA);
  uint4 a1 = *(const uint4*)(gA + 8);
  uint4 b0 = *(const uint4*)(gB);
  uint4 b1 = *(const uint4*)(gB + 8);
  for (int i = 0; i < nt; ++i) {
    __syncthreads();
    *(uint4*)(As + lr * LDT + lc)     = a0;
    *(uint4*)(As + lr * LDT + lc + 8) = a1;
    *(uint4*)(Bs + lr * LDT + lc)     = b0;
    *(uint4*)(Bs + lr * LDT + lc + 8) = b1;
    __syncthreads();
    if (i + 1 < nt) {   // prefetch next tile into registers during compute
      const int kt = (i + 1) << 5;
      a0 = *(const uint4*)(gA + kt);
      a1 = *(const uint4*)(gA + kt + 8);
      b0 = *(const uint4*)(gB + kt);
      b1 = *(const uint4*)(gB + kt + 8);
    }
    v16bf af[2], bfg[4];
#pragma unroll
    for (int ii = 0; ii < 2; ++ii)
      af[ii] = ld_frag(As + (wm * 32 + ii * 16 + rr) * LDT + kb);
#pragma unroll
    for (int j = 0; j < 4; ++j)
      bfg[j] = ld_frag(Bs + (wn * 64 + j * 16 + rr) * LDT + kb);
#pragma unroll
    for (int ii = 0; ii < 2; ++ii)
#pragma unroll
      for (int j = 0; j < 4; ++j)
        acc[ii][j] = wmma_bf16(af[ii], bfg[j], acc[ii][j]);
  }
#endif

#pragma unroll
  for (int i = 0; i < 2; ++i) {
#pragma unroll
    for (int j = 0; j < 4; ++j) {
      const int col = nblk + wn * 64 + j * 16 + rr;
      const float bv = bias[col];
#pragma unroll
      for (int r = 0; r < 8; ++r) {
        const int row = mblk + wm * 32 + i * 16 + hf * 8 + r;
        float v = acc[i][j][r] + bv;
        const size_t o = (size_t)row * N + col;
        if (mode == 0) {
          out_bf[o] = (bf16)v;
        } else if (mode == 1) {
          out_f[o] = resid[o] + gate[(size_t)(row >> 12) * ADA_STRIDE + col] * v;
        } else {
          float t = 0.7978845608028654f * (v + 0.044715f * v * v * v);
          out_bf[o] = (bf16)(0.5f * v * (1.0f + tanhf(t)));
        }
      }
    }
  }
}

// ---------------------------------------------------------------------------
// Windowed attention: one block per (window, head). 128x128 scores,
// softmax, 128x128 output -- all via bf16 WMMA.
// ---------------------------------------------------------------------------
#define ALD 136
__global__ __launch_bounds__(256) void attn_kernel(const bf16* __restrict__ qkv,
                                                   bf16* __restrict__ o) {
  __shared__ bf16 qs[128 * ALD];   // q rows [tok][d]; reused for probs [q][k]
  __shared__ bf16 ks[128 * ALD];   // k rows [tok][d]
  __shared__ bf16 vs[128 * ALD];   // v transposed [d][tok]
  const int tid = threadIdx.x;
  const int lane = tid & 31;
  const int wave = tid >> 5;
  const int hf = lane >> 4;
  const int rr = lane & 15;
  const int kb = hf * 8;
  const int win = blockIdx.x >> 3;
  const int hh = blockIdx.x & 7;
  const int tok0 = win * 128;

  const int r = tid >> 1;
  const int c0 = (tid & 1) * 64;
  const bf16* base = qkv + (size_t)(tok0 + r) * 3072 + hh * 128 + c0;
#pragma unroll
  for (int j = 0; j < 8; ++j) {
    *(uint4*)(qs + r * ALD + c0 + j * 8) = *(const uint4*)(base + j * 8);
    *(uint4*)(ks + r * ALD + c0 + j * 8) = *(const uint4*)(base + 1024 + j * 8);
  }
#pragma unroll
  for (int j = 0; j < 8; ++j) {
    union { uint4 u; bf16 e[8]; } pv;
    pv.u = *(const uint4*)(base + 2048 + j * 8);
#pragma unroll
    for (int e = 0; e < 8; ++e)
      vs[(c0 + j * 8 + e) * ALD + r] = pv.e[e];
  }
  __syncthreads();

  const v8f vzero = {0.f, 0.f, 0.f, 0.f, 0.f, 0.f, 0.f, 0.f};
  v8f sc[8];
#pragma unroll
  for (int j = 0; j < 8; ++j) sc[j] = vzero;
  for (int kk = 0; kk < 128; kk += 32) {
    v16bf aq = ld_frag(qs + (wave * 16 + rr) * ALD + kk + kb);
#pragma unroll
    for (int j = 0; j < 8; ++j) {
      v16bf bk = ld_frag(ks + (j * 16 + rr) * ALD + kk + kb);
      sc[j] = wmma_bf16(aq, bk, sc[j]);
    }
  }
  __syncthreads();   // q tile no longer needed; reuse qs for probs

  const float scl = 0.08838834764831845f;   // 1/sqrt(128)
#pragma unroll
  for (int rI = 0; rI < 8; ++rI) {
    float mx = -3.4e38f;
#pragma unroll
    for (int j = 0; j < 8; ++j) { float v = sc[j][rI] * scl; sc[j][rI] = v; mx = fmaxf(mx, v); }
    mx = fmaxf(mx, __shfl_xor(mx, 1, 32));
    mx = fmaxf(mx, __shfl_xor(mx, 2, 32));
    mx = fmaxf(mx, __shfl_xor(mx, 4, 32));
    mx = fmaxf(mx, __shfl_xor(mx, 8, 32));
    float sum = 0.f;
#pragma unroll
    for (int j = 0; j < 8; ++j) { float e = __expf(sc[j][rI] - mx); sc[j][rI] = e; sum += e; }
    sum += __shfl_xor(sum, 1, 32);
    sum += __shfl_xor(sum, 2, 32);
    sum += __shfl_xor(sum, 4, 32);
    sum += __shfl_xor(sum, 8, 32);
    const float inv = 1.0f / sum;
    const int qrow = wave * 16 + hf * 8 + rI;
#pragma unroll
    for (int j = 0; j < 8; ++j)
      qs[qrow * ALD + j * 16 + rr] = (bf16)(sc[j][rI] * inv);
  }
  __syncthreads();

  v8f oc[8];
#pragma unroll
  for (int j = 0; j < 8; ++j) oc[j] = vzero;
  for (int kk = 0; kk < 128; kk += 32) {
    v16bf ap = ld_frag(qs + (wave * 16 + rr) * ALD + kk + kb);
#pragma unroll
    for (int j = 0; j < 8; ++j) {
      v16bf bv = ld_frag(vs + (j * 16 + rr) * ALD + kk + kb);
      oc[j] = wmma_bf16(ap, bv, oc[j]);
    }
  }
#pragma unroll
  for (int rI = 0; rI < 8; ++rI)
#pragma unroll
    for (int j = 0; j < 8; ++j)
      o[(size_t)(tok0 + wave * 16 + hf * 8 + rI) * 1024 + hh * 128 + j * 16 + rr] =
          (bf16)oc[j][rI];
}

// ---------------------------------------------------------------------------
extern "C" void kernel_launch(void* const* d_in, const int* in_sizes, int n_in,
                              void* d_out, int out_size, void* d_ws, size_t ws_size,
                              hipStream_t stream) {
  (void)in_sizes; (void)n_in; (void)out_size; (void)ws_size;
  const float* group_x = (const float*)d_in[0];
  const float* context = (const float*)d_in[1];
  const float* W_ada  = (const float*)d_in[2];
  const float* b_ada  = (const float*)d_in[3];
  const float* scale1 = (const float*)d_in[4];
  const float* W_qkv  = (const float*)d_in[5];
  const float* b_qkv  = (const float*)d_in[6];
  const float* W_out  = (const float*)d_in[7];
  const float* b_out  = (const float*)d_in[8];
  const float* scale2 = (const float*)d_in[9];
  const float* W_ff1  = (const float*)d_in[10];
  const float* b_ff1  = (const float*)d_in[11];
  const float* W_ff2  = (const float*)d_in[12];
  const float* b_ff2  = (const float*)d_in[13];
  float* out = (float*)d_out;

  char* ws = (char*)d_ws;
  size_t off = 0;
  auto alloc = [&](size_t bytes) {
    char* p = ws + off;
    off += (bytes + 255) & ~(size_t)255;
    return p;
  };
  bf16*  Wqkv_t = (bf16*)alloc(1024ull * 3072 * 2);
  bf16*  Wout_t = (bf16*)alloc(1024ull * 1024 * 2);
  bf16*  Wff1_t = (bf16*)alloc(1024ull * 4096 * 2);
  bf16*  Wff2_t = (bf16*)alloc(4096ull * 1024 * 2);
  float* ada    = (float*)alloc(8ull * 6144 * 4);
  bf16*  xb     = (bf16*)alloc(32768ull * 1024 * 2);
  float* res2   = (float*)alloc(32768ull * 1024 * 4);
  bf16*  qkv    = (bf16*)alloc(32768ull * 3072 * 2);
  bf16*  ob     = (bf16*)alloc(32768ull * 1024 * 2);
  bf16*  h1     = qkv;   // alias: qkv+ob region (268 MB) reused after attention
  bf16*  hmod   = xb;    // alias: xb reused after qkv GEMM

  dim3 blk(256);
  // Weight convert+transpose to bf16 [N][K]
  wconv_kernel<<<dim3(3072 / 256, 1024), blk, 0, stream>>>(W_qkv, Wqkv_t, 1024, 3072);
  wconv_kernel<<<dim3(1024 / 256, 1024), blk, 0, stream>>>(W_out, Wout_t, 1024, 1024);
  wconv_kernel<<<dim3(4096 / 256, 1024), blk, 0, stream>>>(W_ff1, Wff1_t, 1024, 4096);
  wconv_kernel<<<dim3(1024 / 256, 4096), blk, 0, stream>>>(W_ff2, Wff2_t, 4096, 1024);
  // adaLN parameters
  ada_kernel<<<dim3(6144 / 256, 8), blk, 0, stream>>>(context, W_ada, b_ada, ada);
  // norm1 + modulate -> bf16
  rmsnorm_mod_kernel<<<32768, blk, 0, stream>>>(group_x, scale1, ada, 0, 1024, xb);
  // qkv = xb @ Wqkv + b
  gemm_bf16_kernel<<<dim3(3072 / 128, 32768 / 128), blk, 0, stream>>>(
      xb, Wqkv_t, b_qkv, nullptr, nullptr, qkv, nullptr, 3072, 1024, 0);
  // rope(q), rope(k)
  rope_kernel<<<(32768 * 8 * 64) / 256, blk, 0, stream>>>(qkv);
  // windowed attention
  attn_kernel<<<256 * 8, blk, 0, stream>>>(qkv, ob);
  // x = group_x + g_msa * (o @ W_out + b)
  gemm_bf16_kernel<<<dim3(1024 / 128, 32768 / 128), blk, 0, stream>>>(
      ob, Wout_t, b_out, group_x, ada + 2048, nullptr, res2, 1024, 1024, 1);
  // norm2 + modulate -> bf16
  rmsnorm_mod_kernel<<<32768, blk, 0, stream>>>(res2, scale2, ada, 3072, 4096, hmod);
  // h1 = gelu(hmod @ W_ff1 + b)
  gemm_bf16_kernel<<<dim3(4096 / 128, 32768 / 128), blk, 0, stream>>>(
      hmod, Wff1_t, b_ff1, nullptr, nullptr, h1, nullptr, 4096, 1024, 2);
  // out = res2 + g_mlp * (h1 @ W_ff2 + b)
  gemm_bf16_kernel<<<dim3(1024 / 128, 32768 / 128), blk, 0, stream>>>(
      h1, Wff2_t, b_ff2, res2, ada + 5120, nullptr, out, 1024, 4096, 1);
}
